// GRUDecoder_65343632441415
// MI455X (gfx1250) — matrix operand order repaired and checked by reading, not compiled, not run on hardware
//
#include <hip/hip_runtime.h>

// ---------------------------------------------------------------------------
// CDNA5 (gfx1250) implementation of the NRI GRU decoder forward pass.
// All heavy GEMMs run on v_wmma_f32_16x16x32_f16 (f16 in, f32 accumulate).
// ---------------------------------------------------------------------------

typedef __attribute__((ext_vector_type(16))) _Float16 v16h;
typedef __attribute__((ext_vector_type(8)))  _Float16 v8h;
typedef __attribute__((ext_vector_type(8)))  float    v8f;

constexpr int Bc = 4, Nc = 32, Tc = 64, Fc = 4;
constexpr int Ec = Nc * (Nc - 1);       // 992
constexpr int Hc = 256, Kc = 2, D2c = 260;
constexpr int Sc = Tc / 2;              // 32 (pred_steps = 2)
constexpr int RE = Bc * Sc * Ec;        // 126976 edge rows (mult of 16)
constexpr int RN = Bc * Sc * Nc;        // 4096  node rows
constexpr int D2P = 288;                // 260 padded to mult of 32 (K dim)
constexpr int OP260 = 272;              // 260 padded to mult of 16 (out rows)
constexpr float BN_EPS = 1e-5f;

// ---------------------------------------------------------------------------
// WMMA GEMM:  out[r, o] = act( sum_k A[r,k] * W[o,k] + bias[o] )
// A: f16 [R, lda]  (lda multiple of 32, zero padded)
// W: f16 [>=nTiles*16, ldw] (ldw multiple of 32, zero padded)
// EPI: 0 = f16 store, 1 = f16 read-add-write, 2 = f32 store
// ACT: 0 = none, 1 = relu, 2 = elu
// RT : multiply by rel_type[(b*E+e)*K + kk] (row-decoded) before store
// ---------------------------------------------------------------------------
template <int EPI, int ACT, int RT>
__global__ void k_gemm(const _Float16* __restrict__ A, int lda,
                       const _Float16* __restrict__ W, int ldw,
                       const float* __restrict__ bias,
                       _Float16* __restrict__ outH, float* __restrict__ outF,
                       int ldo, int Kdim, int OUT, int nTiles,
                       const float* __restrict__ relt, int kk) {
  const int lane = threadIdx.x & 31;
  const int wv   = threadIdx.x >> 5;
  const int ct   = blockIdx.y * 8 + wv;
  if (ct >= nTiles) return;

  const int r0 = blockIdx.x * 16;
  // A fragment addressing (16x32 f16 A layout, wave32)
  const int arow = r0 + (lane & 15);
  const int ak0  = (lane >> 4) << 4;            // 0 or 16
  // B fragment addressing (32x16 f16 B layout, wave32)
  const int bcol = ct * 16 + (lane & 15);
  const int bk0  = (lane >> 4) << 4;            // 0 or 16

  const _Float16* Ap = A + (size_t)arow * lda + ak0;
  const _Float16* Wp = W + (size_t)bcol * ldw + bk0;

  v8f c = {};
  for (int k0 = 0; k0 < Kdim; k0 += 32) {
    v8h alo = *(const v8h*)(Ap + k0);
    v8h ahi = *(const v8h*)(Ap + k0 + 8);
    v16h a = __builtin_shufflevector(alo, ahi, 0, 1, 2, 3, 4, 5, 6, 7,
                                     8, 9, 10, 11, 12, 13, 14, 15);
    v16h b = *(const v16h*)(Wp + k0);
    c = __builtin_amdgcn_wmma_f32_16x16x32_f16(false, a, false, b,
                                               (short)0, c, false, false);
  }

  const int col = ct * 16 + (lane & 15);
  if (col >= OUT) return;
  const float bs = bias ? bias[col] : 0.0f;
  const int rbase = r0 + ((lane >> 4) << 3);
#pragma unroll
  for (int i = 0; i < 8; ++i) {
    const int row = rbase + i;
    float v = c[i] + bs;
    if (ACT == 1) v = fmaxf(v, 0.0f);
    if (ACT == 2) v = (v > 0.0f) ? v : expm1f(v);
    if (RT) {
      const int bb = row / (Sc * Ec);
      const int ee = row % Ec;
      v *= relt[((size_t)bb * Ec + ee) * Kc + kk];
    }
    const size_t o = (size_t)row * ldo + col;
    if (EPI == 0) outH[o] = (_Float16)v;
    else if (EPI == 1) outH[o] = (_Float16)((float)outH[o] + v);
    else outF[o] = v;
  }
}

// -------------------------- elementwise helpers ----------------------------
__global__ void k_build_idx(const float* __restrict__ rr,
                            const float* __restrict__ rs,
                            int* __restrict__ ridx, int* __restrict__ sidx) {
  int e = blockIdx.x * blockDim.x + threadIdx.x;
  if (e >= Ec) return;
  int r = 0, s = 0;
  for (int n = 0; n < Nc; ++n) {
    if (rr[e * Nc + n] > 0.5f) r = n;
    if (rs[e * Nc + n] > 0.5f) s = n;
  }
  ridx[e] = r;
  sidx[e] = s;
}

__global__ void k_gather_xs(const float* __restrict__ inp,
                            float* __restrict__ xs, int step) {
  int t = blockIdx.x * blockDim.x + threadIdx.x;
  if (t >= RN * Fc) return;
  int f = t % Fc, r = t / Fc;
  int n = r % Nc, bs = r / Nc;
  int s = bs % Sc, b = bs / Sc;
  xs[t] = inp[(((size_t)b * Nc + n) * Tc + (step + 2 * s)) * Fc + f];
}

__global__ void k_build_pre(const float* __restrict__ xs,
                            const int* __restrict__ sidx,
                            const int* __restrict__ ridx,
                            _Float16* __restrict__ pre) {
  int t = blockIdx.x * blockDim.x + threadIdx.x;
  if (t >= RE) return;
  int e = t % Ec, bs = t / Ec;
  const float* xb = xs + (size_t)bs * Nc * Fc;
  size_t o = (size_t)t * 32;
  int sn = sidx[e], rn = ridx[e];
#pragma unroll
  for (int f = 0; f < 4; ++f) pre[o + f] = (_Float16)xb[sn * Fc + f];
#pragma unroll
  for (int f = 0; f < 4; ++f) pre[o + 4 + f] = (_Float16)xb[rn * Fc + f];
#pragma unroll
  for (int cidx = 8; cidx < 32; ++cidx) pre[o + cidx] = (_Float16)0.0f;
}

__global__ void k_cvtw(const float* __restrict__ src, _Float16* __restrict__ dst,
                       int OUT, int IN, int OUTP, int INP) {
  int t = blockIdx.x * blockDim.x + threadIdx.x;
  if (t >= OUTP * INP) return;
  int o = t / INP, i = t % INP;
  dst[t] = (o < OUT && i < IN) ? (_Float16)src[(size_t)o * IN + i]
                               : (_Float16)0.0f;
}

__global__ void k_fill_h(_Float16* __restrict__ p, long n) {
  long t = blockIdx.x * (long)blockDim.x + threadIdx.x;
  if (t < n) p[t] = (_Float16)0.0f;
}

// per-(s, channel) batchnorm stats over (B, inner)
__global__ void k_bn_stats(const _Float16* __restrict__ src, int ld, int C,
                           int inner, float* __restrict__ mean,
                           float* __restrict__ rvar) {
  __shared__ float s1[256], s2[256];
  int s = blockIdx.x / C, cch = blockIdx.x % C;
  int total = Bc * inner;
  float sum = 0.0f, sq = 0.0f;
  for (int t = threadIdx.x; t < total; t += 256) {
    int b = t / inner, i = t % inner;
    float v = (float)src[(size_t)((b * Sc + s) * inner + i) * ld + cch];
    sum += v;
    sq += v * v;
  }
  s1[threadIdx.x] = sum;
  s2[threadIdx.x] = sq;
  __syncthreads();
  for (int st = 128; st > 0; st >>= 1) {
    if (threadIdx.x < st) {
      s1[threadIdx.x] += s1[threadIdx.x + st];
      s2[threadIdx.x] += s2[threadIdx.x + st];
    }
    __syncthreads();
  }
  if (threadIdx.x == 0) {
    float m = s1[0] / total;
    float var = s2[0] / total - m * m;
    mean[s * C + cch] = m;
    rvar[s * C + cch] = rsqrtf(var + BN_EPS);
  }
}

__global__ void k_bn_apply(const _Float16* __restrict__ src,
                           _Float16* __restrict__ dst, int ld, int C, int inner,
                           const float* __restrict__ mean,
                           const float* __restrict__ rvar, long total) {
  long t = blockIdx.x * (long)blockDim.x + threadIdx.x;
  if (t >= total) return;
  long r = t / C;
  int cch = (int)(t % C);
  int s = (int)((r / inner) % Sc);
  size_t o = (size_t)r * ld + cch;
  dst[o] = (_Float16)(((float)src[o] - mean[s * C + cch]) * rvar[s * C + cch]);
}

__global__ void k_gru(const _Float16* __restrict__ xr, const _Float16* __restrict__ hr,
                      const _Float16* __restrict__ xi, const _Float16* __restrict__ hi,
                      const _Float16* __restrict__ xn, const _Float16* __restrict__ hn,
                      _Float16* __restrict__ h, int ld, int C, long total) {
  long t = blockIdx.x * (long)blockDim.x + threadIdx.x;
  if (t >= total) return;
  long r = t / C;
  int cch = (int)(t % C);
  size_t o = (size_t)r * ld + cch;
  float er = 1.0f / (1.0f + __expf(-((float)xr[o] + (float)hr[o])));
  float ei = 1.0f / (1.0f + __expf(-((float)xi[o] + (float)hi[o])));
  float en = tanhf((float)xn[o] + er * (float)hn[o]);
  float ho = (float)h[o];
  h[o] = (_Float16)((1.0f - ei) * en + ei * ho);
}

// agg[b,s,n,h] = sum over the 31 edges whose receiver is n; concat xs; pad 0
__global__ void k_agg_aug(const _Float16* __restrict__ he,
                          const float* __restrict__ xs,
                          _Float16* __restrict__ aug) {
  int t = blockIdx.x * blockDim.x + threadIdx.x;
  if (t >= RN * D2P) return;
  int r = t / D2P, cch = t % D2P;
  float v;
  if (cch < Hc) {
    int n = r % Nc, bs = r / Nc;
    size_t base = ((size_t)bs * Ec + (size_t)n * (Nc - 1)) * Hc + cch;
    float acc = 0.0f;
    for (int q = 0; q < Nc - 1; ++q) acc += (float)he[base + (size_t)q * Hc];
    v = acc;
  } else if (cch < Hc + Fc) {
    v = xs[(size_t)r * Fc + (cch - Hc)];
  } else {
    v = 0.0f;
  }
  aug[(size_t)r * D2P + cch] = (_Float16)v;
}

__global__ void k_pred_out(const float* __restrict__ p3,
                           const float* __restrict__ xs,
                           float* __restrict__ out, int step) {
  int t = blockIdx.x * blockDim.x + threadIdx.x;
  if (t >= RN * Fc) return;
  int f = t % Fc, r = t / Fc;
  int n = r % Nc, bs = r / Nc;
  int s = bs % Sc, b = bs / Sc;
  int tt = step + 2 * s;
  if (tt >= Tc - 1) return;
  out[(((size_t)b * Nc + n) * (Tc - 1) + tt) * Fc + f] = xs[t] + p3[t];
}

// ---------------------------------------------------------------------------
static inline int otiles(int OUT) { return (OUT + 15) / 16; }
static inline dim3 ggrid(int R, int OUT) {
  return dim3(R / 16, (otiles(OUT) + 7) / 8);
}
static inline int blocks(long n) { return (int)((n + 255) / 256); }

extern "C" void kernel_launch(void* const* d_in, const int* in_sizes, int n_in,
                              void* d_out, int out_size, void* d_ws, size_t ws_size,
                              hipStream_t stream) {
  (void)in_sizes; (void)n_in; (void)out_size; (void)ws_size;
  const float* inputs   = (const float*)d_in[0];
  const float* rel_type = (const float*)d_in[1];
  const float* rel_rec  = (const float*)d_in[2];
  const float* rel_send = (const float*)d_in[3];
  const float* msg_W1 = (const float*)d_in[4];
  const float* msg_b1 = (const float*)d_in[5];
  const float* msg_W2 = (const float*)d_in[6];
  const float* msg_b2 = (const float*)d_in[7];
  const float* n2e_W1 = (const float*)d_in[8];
  const float* n2e_b1 = (const float*)d_in[9];
  const float* n2e_W2 = (const float*)d_in[10];
  const float* n2e_b2 = (const float*)d_in[11];
  const float* e2n_W1 = (const float*)d_in[12];
  const float* e2n_b1 = (const float*)d_in[13];
  const float* e2n_W2 = (const float*)d_in[14];
  const float* e2n_b2 = (const float*)d_in[15];
  const float* ge_iW  = (const float*)d_in[16];
  const float* ge_ib  = (const float*)d_in[17];
  const float* ge_hW  = (const float*)d_in[18];
  const float* gn_iW  = (const float*)d_in[19];
  const float* gn_ib  = (const float*)d_in[20];
  const float* gn_hW  = (const float*)d_in[21];
  const float* out_W1 = (const float*)d_in[22];
  const float* out_b1 = (const float*)d_in[23];
  const float* out_W2 = (const float*)d_in[24];
  const float* out_b2 = (const float*)d_in[25];
  const float* out_W3 = (const float*)d_in[26];
  const float* out_b3 = (const float*)d_in[27];
  float* out = (float*)d_out;

  // ---- workspace carve-up (256B aligned) ----
  char* base = (char*)d_ws;
  size_t off = 0;
  auto alloc = [&](size_t bytes) -> void* {
    void* p = base + off;
    off = (off + bytes + 255) & ~(size_t)255;
    return p;
  };
  const size_t REH = (size_t)RE * Hc;
  _Float16* e_pre = (_Float16*)alloc((size_t)RE * 32 * 2);
  _Float16* e_t[6];
  for (int i = 0; i < 6; ++i) e_t[i] = (_Float16*)alloc(REH * 2);
  _Float16* e_am = (_Float16*)alloc(REH * 2);
  _Float16* e_he = (_Float16*)alloc(REH * 2);
  // node f16 tensors in one block so one fill covers all pads
  const size_t NODEN = (size_t)RN * D2P;
  _Float16* n_all = (_Float16*)alloc(8 * NODEN * 2);
  _Float16* n_aug = n_all;
  _Float16* n_t[6];
  for (int i = 0; i < 6; ++i) n_t[i] = n_all + (1 + i) * NODEN;
  _Float16* n_hn = n_all + 7 * NODEN;
  _Float16* n_p1 = (_Float16*)alloc((size_t)RN * 256 * 2);
  _Float16* n_p2 = (_Float16*)alloc((size_t)RN * 256 * 2);
  float* n_p3 = (float*)alloc((size_t)RN * 4 * 4);
  float* xs   = (float*)alloc((size_t)RN * Fc * 4);
  int* sidx = (int*)alloc(Ec * 4);
  int* ridx = (int*)alloc(Ec * 4);
  float* bmean = (float*)alloc((size_t)Sc * D2c * 4);
  float* brvar = (float*)alloc((size_t)Sc * D2c * 4);
  // f16 weights (padded)
  _Float16* msgW1h = (_Float16*)alloc((size_t)Kc * 256 * 32 * 2);
  _Float16* msgW2h = (_Float16*)alloc((size_t)Kc * 256 * 256 * 2);
  _Float16* n2e1h  = (_Float16*)alloc((size_t)256 * 256 * 2);
  _Float16* n2e2h  = (_Float16*)alloc((size_t)256 * 256 * 2);
  _Float16* geIh   = (_Float16*)alloc((size_t)3 * 256 * 256 * 2);
  _Float16* geHh   = (_Float16*)alloc((size_t)3 * 256 * 256 * 2);
  _Float16* e2n1h  = (_Float16*)alloc((size_t)OP260 * D2P * 2);
  _Float16* e2n2h  = (_Float16*)alloc((size_t)OP260 * D2P * 2);
  _Float16* gnIh   = (_Float16*)alloc((size_t)3 * OP260 * D2P * 2);
  _Float16* gnHh   = (_Float16*)alloc((size_t)3 * OP260 * D2P * 2);
  _Float16* ow1h   = (_Float16*)alloc((size_t)256 * D2P * 2);
  _Float16* ow2h   = (_Float16*)alloc((size_t)256 * 256 * 2);
  _Float16* ow3h   = (_Float16*)alloc((size_t)16 * 256 * 2);

  // ---- one-time-per-launch prep ----
  k_fill_h<<<blocks(8 * (long)NODEN), 256, 0, stream>>>(n_all, 8 * (long)NODEN);
  k_build_idx<<<blocks(Ec), 256, 0, stream>>>(rel_rec, rel_send, ridx, sidx);
  auto cvt = [&](const float* s, _Float16* d, int O, int I, int OP, int IP) {
    k_cvtw<<<blocks((long)OP * IP), 256, 0, stream>>>(s, d, O, I, OP, IP);
  };
  for (int k = 0; k < Kc; ++k) {
    cvt(msg_W1 + (size_t)k * Hc * 8,  msgW1h + (size_t)k * 256 * 32, 256, 8, 256, 32);
    cvt(msg_W2 + (size_t)k * Hc * Hc, msgW2h + (size_t)k * 256 * 256, 256, 256, 256, 256);
  }
  cvt(n2e_W1, n2e1h, 256, 256, 256, 256);
  cvt(n2e_W2, n2e2h, 256, 256, 256, 256);
  for (int g = 0; g < 3; ++g) {
    cvt(ge_iW + (size_t)g * Hc * Hc, geIh + (size_t)g * 256 * 256, 256, 256, 256, 256);
    cvt(ge_hW + (size_t)g * Hc * Hc, geHh + (size_t)g * 256 * 256, 256, 256, 256, 256);
    cvt(gn_iW + (size_t)g * D2c * D2c, gnIh + (size_t)g * OP260 * D2P, 260, 260, OP260, D2P);
    cvt(gn_hW + (size_t)g * D2c * D2c, gnHh + (size_t)g * OP260 * D2P, 260, 260, OP260, D2P);
  }
  cvt(e2n_W1, e2n1h, 260, 260, OP260, D2P);
  cvt(e2n_W2, e2n2h, 260, 260, OP260, D2P);
  cvt(out_W1, ow1h, 256, 260, 256, D2P);
  cvt(out_W2, ow2h, 256, 256, 256, 256);
  cvt(out_W3, ow3h, 4, 256, 16, 256);

  // ---- two prediction steps ----
  for (int step = 0; step < 2; ++step) {
    k_gather_xs<<<blocks((long)RN * Fc), 256, 0, stream>>>(inputs, xs, step);
    k_build_pre<<<blocks(RE), 256, 0, stream>>>(xs, sidx, ridx, e_pre);

    // messages: am = sum_k relu(relu(pre W1k + b1k) W2k + b2k) * rel_type[...,k]
    for (int k = 0; k < Kc; ++k) {
      k_gemm<0, 1, 0><<<ggrid(RE, 256), 256, 0, stream>>>(
          e_pre, 32, msgW1h + (size_t)k * 256 * 32, 32, msg_b1 + k * Hc,
          e_t[0], nullptr, 256, 32, 256, otiles(256), nullptr, 0);
      if (k == 0)
        k_gemm<0, 1, 1><<<ggrid(RE, 256), 256, 0, stream>>>(
            e_t[0], 256, msgW2h, 256, msg_b2, e_am, nullptr, 256, 256, 256,
            otiles(256), rel_type, 0);
      else
        k_gemm<1, 1, 1><<<ggrid(RE, 256), 256, 0, stream>>>(
            e_t[0], 256, msgW2h + (size_t)k * 256 * 256, 256, msg_b2 + k * Hc,
            e_am, nullptr, 256, 256, 256, otiles(256), rel_type, k);
    }

    if (step == 0) {
      // h_e = batchnorm(elu(elu(am W1 + b1) W2 + b2))
      k_gemm<0, 2, 0><<<ggrid(RE, 256), 256, 0, stream>>>(
          e_am, 256, n2e1h, 256, n2e_b1, e_t[0], nullptr, 256, 256, 256,
          otiles(256), nullptr, 0);
      k_gemm<0, 2, 0><<<ggrid(RE, 256), 256, 0, stream>>>(
          e_t[0], 256, n2e2h, 256, n2e_b2, e_t[1], nullptr, 256, 256, 256,
          otiles(256), nullptr, 0);
      k_bn_stats<<<Sc * Hc, 256, 0, stream>>>(e_t[1], 256, Hc, Ec, bmean, brvar);
      k_bn_apply<<<blocks((long)RE * Hc), 256, 0, stream>>>(
          e_t[1], e_he, 256, Hc, Ec, bmean, brvar, (long)RE * Hc);
    } else {
      // edge GRU
      for (int g = 0; g < 3; ++g)
        k_gemm<0, 0, 0><<<ggrid(RE, 256), 256, 0, stream>>>(
            e_he, 256, geHh + (size_t)g * 256 * 256, 256, nullptr,
            e_t[1 + g], nullptr, 256, 256, 256, otiles(256), nullptr, 0);
      k_gemm<0, 0, 0><<<ggrid(RE, 256), 256, 0, stream>>>(
          e_am, 256, geIh, 256, ge_ib, e_t[4], nullptr, 256, 256, 256,
          otiles(256), nullptr, 0);
      k_gemm<0, 0, 0><<<ggrid(RE, 256), 256, 0, stream>>>(
          e_am, 256, geIh + (size_t)1 * 256 * 256, 256, ge_ib + Hc, e_t[5],
          nullptr, 256, 256, 256, otiles(256), nullptr, 0);
      k_gemm<0, 0, 0><<<ggrid(RE, 256), 256, 0, stream>>>(
          e_am, 256, geIh + (size_t)2 * 256 * 256, 256, ge_ib + 2 * Hc, e_t[0],
          nullptr, 256, 256, 256, otiles(256), nullptr, 0);
      k_gru<<<blocks((long)RE * Hc), 256, 0, stream>>>(
          e_t[4], e_t[1], e_t[5], e_t[2], e_t[0], e_t[3], e_he, 256, Hc,
          (long)RE * Hc);
    }

    // aggregate to nodes + concat xs
    k_agg_aug<<<blocks((long)RN * D2P), 256, 0, stream>>>(e_he, xs, n_aug);

    if (step == 0) {
      k_gemm<0, 2, 0><<<ggrid(RN, 260), 256, 0, stream>>>(
          n_aug, D2P, e2n1h, D2P, e2n_b1, n_t[0], nullptr, D2P, D2P, 260,
          otiles(260), nullptr, 0);
      k_gemm<0, 2, 0><<<ggrid(RN, 260), 256, 0, stream>>>(
          n_t[0], D2P, e2n2h, D2P, e2n_b2, n_t[1], nullptr, D2P, D2P, 260,
          otiles(260), nullptr, 0);
      k_bn_stats<<<Sc * D2c, 256, 0, stream>>>(n_t[1], D2P, D2c, Nc, bmean, brvar);
      k_bn_apply<<<blocks((long)RN * D2c), 256, 0, stream>>>(
          n_t[1], n_hn, D2P, D2c, Nc, bmean, brvar, (long)RN * D2c);
    } else {
      // node GRU
      for (int g = 0; g < 3; ++g)
        k_gemm<0, 0, 0><<<ggrid(RN, 260), 256, 0, stream>>>(
            n_hn, D2P, gnHh + (size_t)g * OP260 * D2P, D2P, nullptr, n_t[g],
            nullptr, D2P, D2P, 260, otiles(260), nullptr, 0);
      for (int g = 0; g < 3; ++g)
        k_gemm<0, 0, 0><<<ggrid(RN, 260), 256, 0, stream>>>(
            n_aug, D2P, gnIh + (size_t)g * OP260 * D2P, D2P, gn_ib + g * D2c,
            n_t[3 + g], nullptr, D2P, D2P, 260, otiles(260), nullptr, 0);
      k_gru<<<blocks((long)RN * D2c), 256, 0, stream>>>(
          n_t[3], n_t[0], n_t[4], n_t[1], n_t[5], n_t[2], n_hn, D2P, D2c,
          (long)RN * D2c);
    }

    // output MLP + residual scatter
    k_gemm<0, 1, 0><<<ggrid(RN, 256), 256, 0, stream>>>(
        n_hn, D2P, ow1h, D2P, out_b1, n_p1, nullptr, 256, D2P, 256,
        otiles(256), nullptr, 0);
    k_gemm<0, 1, 0><<<ggrid(RN, 256), 256, 0, stream>>>(
        n_p1, 256, ow2h, 256, out_b2, n_p2, nullptr, 256, 256, 256,
        otiles(256), nullptr, 0);
    k_gemm<2, 0, 0><<<ggrid(RN, 4), 256, 0, stream>>>(
        n_p2, 256, ow3h, 256, out_b3, nullptr, n_p3, 4, 256, 4, otiles(4),
        nullptr, 0);
    k_pred_out<<<blocks((long)RN * Fc), 256, 0, stream>>>(n_p3, xs, out, step);
  }
}
